// EncodeProcessDecode_84507776516378
// MI455X (gfx1250) — matrix operand order repaired and checked by reading, not compiled, not run on hardware
//
#include <hip/hip_runtime.h>

#define NND   8192
#define DIN   12
#define LATD  128
#define OUTD  3
#define NEDGE 65536
#define NSTEP 15

typedef __attribute__((ext_vector_type(16))) _Float16 v16h;
typedef __attribute__((ext_vector_type(8)))  _Float16 v8h;
typedef __attribute__((ext_vector_type(8)))  float    v8f;

// ---------------- utility kernels ----------------

__global__ void k_zero_u32(unsigned* __restrict__ p, int n) {
    int i = blockIdx.x * blockDim.x + threadIdx.x;
    int stride = gridDim.x * blockDim.x;
    for (; i < n; i += stride) p[i] = 0u;
}

__global__ void k_cvt_f32_f16(const float* __restrict__ x, _Float16* __restrict__ y, int n) {
    int i = blockIdx.x * blockDim.x + threadIdx.x;
    int stride = gridDim.x * blockDim.x;
    for (; i < n; i += stride) y[i] = (_Float16)x[i];
}

// Edge dedup: adjacency is built with .set (duplicates count once).
__global__ void k_mark(const int* __restrict__ s, const int* __restrict__ r,
                       unsigned* __restrict__ bm, int* __restrict__ keep) {
    int e = blockIdx.x * blockDim.x + threadIdx.x;
    if (e >= NEDGE) return;
    unsigned long long pos = (unsigned long long)s[e] * (unsigned long long)NND + (unsigned)r[e];
    unsigned w = (unsigned)(pos >> 5), bit = (unsigned)(pos & 31u);
    unsigned old = atomicOr(&bm[w], 1u << bit);
    keep[e] = ((old >> bit) & 1u) ? 0 : 1;
}

// fp32 W[mat][k][n] -> fp16 Wt[mat][n][k]  (transposed so WMMA B-frags are contiguous)
__global__ void k_wt(const float* __restrict__ W, _Float16* __restrict__ Wt) {
    int mat = blockIdx.y;
    int idx = blockIdx.x * blockDim.x + threadIdx.x;   // 0..16383
    int n = idx >> 7, k = idx & 127;
    Wt[(size_t)mat * (LATD * LATD) + idx] =
        (_Float16)W[(size_t)mat * (LATD * LATD) + (size_t)k * LATD + n];
}

// ---------------- encoder first layer (K=12, scalar) ----------------
__global__ void k_enc0(const float* __restrict__ nfeat, const float* __restrict__ W,
                       const float* __restrict__ b, _Float16* __restrict__ out) {
    int idx = blockIdx.x * blockDim.x + threadIdx.x;   // N*128
    int row = idx >> 7, col = idx & 127;
    float s = b[col];
#pragma unroll
    for (int k = 0; k < DIN; ++k) s += nfeat[row * DIN + k] * W[k * LATD + col];
    out[idx] = (_Float16)fmaxf(s, 0.f);
}

// ---------------- WMMA GEMM: Y[8192,128] = act(X @ W + b) ----------------
// X: f16 row-major [rows][128]. Wt: f16 [n][k]. 256 threads = 8 waves, 128 rows/WG.
__global__ __launch_bounds__(256)
void k_gemm128(const _Float16* __restrict__ X, const _Float16* __restrict__ Wt,
               const float* __restrict__ bias, float* __restrict__ Yf,
               _Float16* __restrict__ Yh, int relu) {
    __shared__ alignas(64) _Float16 wl[LATD * LATD];   // 32 KB weight tile

    // Stage W into LDS via the CDNA5 async DMA path (no VGPR round-trip).
    // VDST VGPR carries the LDS byte address; low 32 bits of a generic
    // pointer to a __shared__ object are exactly that address.
    {
#pragma unroll
        for (int j = 0; j < (LATD * LATD * 2) / (16 * 256); ++j) {   // 8 iters
            const int i = j * 256 + threadIdx.x;                      // 16B chunk id
            const unsigned lds = (unsigned)(size_t)(wl + (size_t)i * 8);
            const _Float16* g = Wt + (size_t)i * 8;
            asm volatile("global_load_async_to_lds_b128 %0, %1, off"
                         :: "v"(lds), "v"(g) : "memory");
        }
        asm volatile("s_wait_asynccnt 0" ::: "memory");   // this wave's DMAs done
    }
    __syncthreads();                                      // publish across waves

    const int wave = threadIdx.x >> 5;
    const int lane = threadIdx.x & 31;
    const int half = lane >> 4;       // lane group 0/1
    const int lm   = lane & 15;       // M (A rows) / N (B,C cols) index
    const int row0 = blockIdx.x * 128 + wave * 16;
    const _Float16* arow = X + (size_t)(row0 + lm) * LATD;

    const v8f vzero = {0.f, 0.f, 0.f, 0.f, 0.f, 0.f, 0.f, 0.f};
    v8f acc[8];
#pragma unroll
    for (int i = 0; i < 8; ++i) acc[i] = vzero;

#pragma unroll
    for (int k0 = 0; k0 < LATD; k0 += 32) {
        if (k0 + 32 < LATD) __builtin_prefetch(arow + k0 + 32, 0, 3);  // global_prefetch_b8
        // A 16x32 f16 frag: lane half h -> K k0 + h*16 + 0..15 (contiguous)
        union { v16h v; v8h h[2]; } a;
        a.v = *(const v16h*)(arow + k0 + half * 16);
#pragma unroll
        for (int nt = 0; nt < 8; ++nt) {
            // B 32x16 f16 frag: lane holds col n = nt*16+lm, K = k0 + half*16 + 0..15 (contiguous in Wt)
            const v16h b = *(const v16h*)(wl + (size_t)(nt * 16 + lm) * LATD + k0 + half * 16);
            acc[nt] = __builtin_amdgcn_wmma_f32_16x16x32_f16(
                false, a.v, false, b, (short)0, acc[nt], false, false);
        }
    }

#pragma unroll
    for (int nt = 0; nt < 8; ++nt) {
        const int ng = nt * 16 + lm;
        const float bv = bias ? bias[ng] : 0.f;
#pragma unroll
        for (int r = 0; r < 8; ++r) {
            float y = acc[nt][r] + bv;
            if (relu) y = fmaxf(y, 0.f);
            const size_t o = (size_t)(row0 + r + 8 * half) * LATD + ng;   // D: VGPR r -> row r+8*half
            if (Yf) Yf[o] = y;
            if (Yh) Yh[o] = (_Float16)y;
        }
    }
}

// ---------------- edge aggregation: feat[r] += v[s] over unique edges ----------------
__global__ __launch_bounds__(128)
void k_scatter(const float* __restrict__ v, const int* __restrict__ s,
               const int* __restrict__ r, const int* __restrict__ keep,
               float* __restrict__ feat) {
    const int e = blockIdx.x;
    if (!keep[e]) return;
    const int c = threadIdx.x;
    atomicAdd(&feat[(size_t)r[e] * LATD + c], v[(size_t)s[e] * LATD + c]);
}

// ---------------- LayerNorm (+optional residual): nf = [nf +] LN(X)*g + be ----------------
__global__ __launch_bounds__(128)
void k_ln(const float* __restrict__ X, const float* __restrict__ g,
          const float* __restrict__ be, float* __restrict__ nf,
          _Float16* __restrict__ nf_h, int residual) {
    __shared__ float sbuf[128];
    const int row = blockIdx.x, tid = threadIdx.x;
    const size_t off = (size_t)row * LATD + tid;
    const float x = X[off];
    sbuf[tid] = x;
    __syncthreads();
    for (int s = 64; s > 0; s >>= 1) { if (tid < s) sbuf[tid] += sbuf[tid + s]; __syncthreads(); }
    const float mean = sbuf[0] * (1.f / 128.f);
    __syncthreads();
    const float d = x - mean;
    sbuf[tid] = d * d;
    __syncthreads();
    for (int s = 64; s > 0; s >>= 1) { if (tid < s) sbuf[tid] += sbuf[tid + s]; __syncthreads(); }
    const float var = sbuf[0] * (1.f / 128.f);
    float y = d * rsqrtf(var + 1e-5f) * g[tid] + be[tid];
    if (residual) y += nf[off];
    nf[off] = y;
    nf_h[off] = (_Float16)y;
}

// ---------------- decoder last layer (N=3, scalar) ----------------
__global__ void k_dec2(const _Float16* __restrict__ X, const float* __restrict__ W,
                       const float* __restrict__ b, float* __restrict__ out) {
    int idx = blockIdx.x * blockDim.x + threadIdx.x;   // N*3
    if (idx >= NND * OUTD) return;
    int row = idx / OUTD, c = idx % OUTD;
    float s = b[c];
#pragma unroll 16
    for (int k = 0; k < LATD; ++k) s += (float)X[(size_t)row * LATD + k] * W[k * OUTD + c];
    out[idx] = s;
}

// ---------------- host orchestration ----------------
extern "C" void kernel_launch(void* const* d_in, const int* in_sizes, int n_in,
                              void* d_out, int out_size, void* d_ws, size_t ws_size,
                              hipStream_t stream) {
    const float* node_features = (const float*)d_in[0];
    const int*   senders   = (const int*)d_in[1];
    const int*   recievers = (const int*)d_in[2];
    const float* enc_W0 = (const float*)d_in[3];
    const float* enc_b0 = (const float*)d_in[4];
    const float* enc_W1 = (const float*)d_in[5];
    const float* enc_b1 = (const float*)d_in[6];
    const float* enc_W2 = (const float*)d_in[7];
    const float* enc_b2 = (const float*)d_in[8];
    const float* enc_g  = (const float*)d_in[9];
    const float* enc_be = (const float*)d_in[10];
    const float* blk_Wv = (const float*)d_in[11];
    const float* blk_W0 = (const float*)d_in[12];
    const float* blk_b0 = (const float*)d_in[13];
    const float* blk_W1 = (const float*)d_in[14];
    const float* blk_b1 = (const float*)d_in[15];
    const float* blk_W2 = (const float*)d_in[16];
    const float* blk_b2 = (const float*)d_in[17];
    const float* blk_g  = (const float*)d_in[18];
    const float* blk_be = (const float*)d_in[19];
    const float* dec_W0 = (const float*)d_in[20];
    const float* dec_b0 = (const float*)d_in[21];
    const float* dec_W1 = (const float*)d_in[22];
    const float* dec_b1 = (const float*)d_in[23];
    const float* dec_W2 = (const float*)d_in[24];
    const float* dec_b2 = (const float*)d_in[25];
    float* out = (float*)d_out;

    char* base = (char*)d_ws;
    size_t off = 0;
    auto take = [&](size_t bytes) -> void* {
        void* p = base + off;
        off += (bytes + 255) & ~(size_t)255;
        return p;
    };
    const size_t NL = (size_t)NND * LATD;
    float*    nf     = (float*)take(NL * 4);
    float*    h3     = (float*)take(NL * 4);       // pre-LN buffer
    float*    vbuf   = (float*)take(NL * 4);       // value projection
    float*    feat   = (float*)take(NL * 4);       // aggregated
    _Float16* nf_h   = (_Float16*)take(NL * 2);
    _Float16* feat_h = (_Float16*)take(NL * 2);
    _Float16* h1h    = (_Float16*)take(NL * 2);
    _Float16* h2h    = (_Float16*)take(NL * 2);
    const size_t MB = (size_t)LATD * LATD;         // one 128x128 matrix
    _Float16* encW1t = (_Float16*)take(MB * 2);
    _Float16* encW2t = (_Float16*)take(MB * 2);
    _Float16* decW0t = (_Float16*)take(MB * 2);
    _Float16* decW1t = (_Float16*)take(MB * 2);
    _Float16* Wvt    = (_Float16*)take((size_t)NSTEP * MB * 2);
    _Float16* W0t    = (_Float16*)take((size_t)NSTEP * MB * 2);
    _Float16* W1t    = (_Float16*)take((size_t)NSTEP * MB * 2);
    _Float16* W2t    = (_Float16*)take((size_t)NSTEP * MB * 2);
    unsigned* bm     = (unsigned*)take((size_t)NND * NND / 8);
    int*      keep   = (int*)take((size_t)NEDGE * 4);

    // --- edge dedup (matches .set() semantics of dense adjacency build) ---
    k_zero_u32<<<2048, 256, 0, stream>>>(bm, NND * NND / 32);
    k_mark<<<NEDGE / 256, 256, 0, stream>>>(senders, recievers, bm, keep);

    // --- convert + transpose all 128x128 weights to f16 [n][k] ---
    k_wt<<<dim3(64, 1), 256, 0, stream>>>(enc_W1, encW1t);
    k_wt<<<dim3(64, 1), 256, 0, stream>>>(enc_W2, encW2t);
    k_wt<<<dim3(64, 1), 256, 0, stream>>>(dec_W0, decW0t);
    k_wt<<<dim3(64, 1), 256, 0, stream>>>(dec_W1, decW1t);
    k_wt<<<dim3(64, NSTEP), 256, 0, stream>>>(blk_Wv, Wvt);
    k_wt<<<dim3(64, NSTEP), 256, 0, stream>>>(blk_W0, W0t);
    k_wt<<<dim3(64, NSTEP), 256, 0, stream>>>(blk_W1, W1t);
    k_wt<<<dim3(64, NSTEP), 256, 0, stream>>>(blk_W2, W2t);

    // --- encoder ---
    k_enc0<<<(NND * LATD) / 256, 256, 0, stream>>>(node_features, enc_W0, enc_b0, h1h);
    k_gemm128<<<NND / 128, 256, 0, stream>>>(h1h, encW1t, enc_b1, nullptr, h2h, 1);
    k_gemm128<<<NND / 128, 256, 0, stream>>>(h2h, encW2t, enc_b2, h3, nullptr, 0);
    k_ln<<<NND, 128, 0, stream>>>(h3, enc_g, enc_be, nf, nf_h, 0);

    // --- 15 processor blocks ---
    for (int t = 0; t < NSTEP; ++t) {
        k_zero_u32<<<1024, 256, 0, stream>>>((unsigned*)feat, (int)NL);
        k_gemm128<<<NND / 128, 256, 0, stream>>>(nf_h, Wvt + (size_t)t * MB, nullptr,
                                                 vbuf, nullptr, 0);
        k_scatter<<<NEDGE, 128, 0, stream>>>(vbuf, senders, recievers, keep, feat);
        k_cvt_f32_f16<<<1024, 256, 0, stream>>>(feat, feat_h, (int)NL);
        k_gemm128<<<NND / 128, 256, 0, stream>>>(feat_h, W0t + (size_t)t * MB,
                                                 blk_b0 + t * LATD, nullptr, h1h, 1);
        k_gemm128<<<NND / 128, 256, 0, stream>>>(h1h, W1t + (size_t)t * MB,
                                                 blk_b1 + t * LATD, nullptr, h2h, 1);
        k_gemm128<<<NND / 128, 256, 0, stream>>>(h2h, W2t + (size_t)t * MB,
                                                 blk_b2 + t * LATD, h3, nullptr, 0);
        k_ln<<<NND, 128, 0, stream>>>(h3, blk_g + t * LATD, blk_be + t * LATD, nf, nf_h, 1);
    }

    // --- decoder ---
    k_gemm128<<<NND / 128, 256, 0, stream>>>(nf_h, decW0t, dec_b0, nullptr, h1h, 1);
    k_gemm128<<<NND / 128, 256, 0, stream>>>(h1h, decW1t, dec_b1, nullptr, h2h, 1);
    k_dec2<<<(NND * OUTD + 255) / 256, 256, 0, stream>>>(h2h, dec_W2, dec_b2, out);
}
